// TransformerBlock_31705448579429
// MI455X (gfx1250) — matrix operand order repaired and checked by reading, not compiled, not run on hardware
//
#include <hip/hip_runtime.h>
#include <hip/hip_bf16.h>
#include <cstdint>
#include <cstddef>

#define Dm   1024
#define Tm   2048
#define Hm   16
#define HDm  64
#define Em   8
#define FFm  2816
#define ROWSm 4608   // 2*Tm + Em*64 padding capacity for expert-compacted rows

typedef __attribute__((ext_vector_type(16))) __bf16         v16bf;
typedef __attribute__((ext_vector_type(16))) unsigned short v16us;
typedef __attribute__((ext_vector_type(8)))  float          v8f;

// ---------- scalar helpers ----------
__device__ __forceinline__ unsigned short f2bf(float f) {
  unsigned u = __float_as_uint(f);
  u += 0x7fffu + ((u >> 16) & 1u);          // round-to-nearest-even
  return (unsigned short)(u >> 16);
}
__device__ __forceinline__ float bf2f(unsigned short h) {
  return __uint_as_float(((unsigned)h) << 16);
}
__device__ __forceinline__ v8f zero8() {
  v8f z;
#pragma unroll
  for (int i = 0; i < 8; ++i) z[i] = 0.f;
  return z;
}
__device__ __forceinline__ v8f wmma_bf16(v16bf a, v16bf b, v8f c) {
  return __builtin_amdgcn_wmma_f32_16x16x32_bf16(false, a, false, b, (short)0, c,
                                                 false, false);
}

// ---------- CDNA5 async global->LDS staging (ASYNCcnt-tracked DMA path) ----------
__device__ __forceinline__ void async_g2l_b128(void* lds, const void* gaddr) {
  unsigned l = (unsigned)(unsigned long long)lds;   // low 32 bits = LDS offset
  asm volatile("global_load_async_to_lds_b128 %0, %1, off"
               :: "v"(l), "v"(gaddr) : "memory");
}
#define WAIT_ASYNC(n) asm volatile("s_wait_asynccnt " #n ::: "memory")
__device__ __forceinline__ void wait_ds0() {
  asm volatile("s_wait_dscnt 0" ::: "memory");
}

// ---------- WMMA fragment loaders (layouts per CDNA5 ISA) ----------
// A fragment: 16x32 (MxK), row-major source; two contiguous 8-elem runs per lane
// -> 2x ds_load_b128.
__device__ __forceinline__ v16bf load_a_frag(const unsigned short* A, int ld) {
  const int lane = threadIdx.x & 31;
  const int half = lane >> 4, m = lane & 15;
  const unsigned short* row = A + m * ld;
  v16us u;
#pragma unroll
  for (int v = 0; v < 8; ++v) {
    int kb = (v < 4) ? (half * 8 + 2 * v) : (16 + half * 8 + 2 * (v - 4));
    u[2 * v]     = row[kb];
    u[2 * v + 1] = row[kb + 1];
  }
  return __builtin_bit_cast(v16bf, u);
}
// Same but f32 source converted on the fly (P tiles in attention).
__device__ __forceinline__ v16bf load_a_frag_f32(const float* A, int ld) {
  const int lane = threadIdx.x & 31;
  const int half = lane >> 4, m = lane & 15;
  const float* row = A + m * ld;
  v16us u;
#pragma unroll
  for (int v = 0; v < 8; ++v) {
    int kb = (v < 4) ? (half * 8 + 2 * v) : (16 + half * 8 + 2 * (v - 4));
    u[2 * v]     = f2bf(row[kb]);
    u[2 * v + 1] = f2bf(row[kb + 1]);
  }
  return __builtin_bit_cast(v16bf, u);
}
// B fragment from (K,N) storage: lane%16 = N, elements over K in table order.
__device__ __forceinline__ v16bf load_bN_frag(const unsigned short* B, int ld) {
  const int lane = threadIdx.x & 31;
  const int half = lane >> 4, n = lane & 15;
  v16us u;
#pragma unroll
  for (int e = 0; e < 16; ++e) {
    int k = (e < 8) ? (half * 8 + e) : (16 + half * 8 + (e - 8));
    u[e] = B[k * ld + n];
  }
  return __builtin_bit_cast(v16bf, u);
}

// ---------- small utility kernels ----------
__global__ void tb_zero_ints(int* p, int n) {
  int i = blockIdx.x * blockDim.x + threadIdx.x;
  if (i < n) p[i] = 0;
}

__global__ __launch_bounds__(256) void tb_cast8(const float* __restrict__ in,
                                                unsigned short* __restrict__ out,
                                                size_t n) {
  size_t i = ((size_t)blockIdx.x * 256 + threadIdx.x) * 8;
  if (i >= n) return;
  float4 a = *(const float4*)&in[i];
  float4 b = *(const float4*)&in[i + 4];
  unsigned r0 = (unsigned)f2bf(a.x) | ((unsigned)f2bf(a.y) << 16);
  unsigned r1 = (unsigned)f2bf(a.z) | ((unsigned)f2bf(a.w) << 16);
  unsigned r2 = (unsigned)f2bf(b.x) | ((unsigned)f2bf(b.y) << 16);
  unsigned r3 = (unsigned)f2bf(b.z) | ((unsigned)f2bf(b.w) << 16);
  uint4 pk; pk.x = r0; pk.y = r1; pk.z = r2; pk.w = r3;
  *(uint4*)&out[i] = pk;
}

// f32 (R,C) -> bf16 (C,R) transposed cast, 32x32 LDS tiles, z = matrix index
__global__ __launch_bounds__(256) void tb_castT(const float* __restrict__ in,
                                                unsigned short* __restrict__ outT,
                                                int R, int C) {
  const size_t base = (size_t)blockIdx.z * R * C;
  const int c0 = blockIdx.x * 32, r0 = blockIdx.y * 32;
  __shared__ unsigned short tile[32][36];
  const int r = threadIdx.x >> 3, c = (threadIdx.x & 7) * 4;
  float4 v = *(const float4*)&in[base + (size_t)(r0 + r) * C + c0 + c];
  tile[r][c]     = f2bf(v.x);
  tile[r][c + 1] = f2bf(v.y);
  tile[r][c + 2] = f2bf(v.z);
  tile[r][c + 3] = f2bf(v.w);
  __syncthreads();
  unsigned p0 = (unsigned)tile[c + 0][r] | ((unsigned)tile[c + 1][r] << 16);
  unsigned p1 = (unsigned)tile[c + 2][r] | ((unsigned)tile[c + 3][r] << 16);
  uint2 pk; pk.x = p0; pk.y = p1;
  *(uint2*)&outT[base + (size_t)(c0 + r) * R + r0 + c] = pk;
}

// bf16 K transpose: (b,s,h,hd) -> kT[(b*H+h)*HD + hd][s]
__global__ __launch_bounds__(256) void tb_ktrans(const unsigned short* __restrict__ kin,
                                                 unsigned short* __restrict__ kT) {
  const int bh = blockIdx.z;               // b*16 + h
  const int b = bh >> 4, h = bh & 15;
  const int s0 = blockIdx.x * 32, hd0 = blockIdx.y * 32;
  __shared__ unsigned short tile[32][36];  // [s][hd]
  const int r = threadIdx.x >> 3, c = (threadIdx.x & 7) * 4;
  uint2 v = *(const uint2*)&kin[(size_t)(b * 1024 + s0 + r) * Dm + h * HDm + hd0 + c];
  tile[r][c]     = (unsigned short)(v.x & 0xffffu);
  tile[r][c + 1] = (unsigned short)(v.x >> 16);
  tile[r][c + 2] = (unsigned short)(v.y & 0xffffu);
  tile[r][c + 3] = (unsigned short)(v.y >> 16);
  __syncthreads();
  unsigned p0 = (unsigned)tile[c + 0][r] | ((unsigned)tile[c + 1][r] << 16);
  unsigned p1 = (unsigned)tile[c + 2][r] | ((unsigned)tile[c + 3][r] << 16);
  uint2 pk; pk.x = p0; pk.y = p1;
  *(uint2*)&kT[(size_t)(bh * HDm + hd0 + r) * 1024 + s0 + c] = pk;
}

// rmsnorm over D=1024, one block per token, outputs bf16
__global__ __launch_bounds__(256) void tb_rmsnorm_cast(const float* __restrict__ x,
                                                       const float* __restrict__ w,
                                                       unsigned short* __restrict__ out) {
  const int t = blockIdx.x, tid = threadIdx.x;
  const int lane = tid & 31, wid = tid >> 5;
  const float* xr = x + (size_t)t * Dm;
  const int i0 = tid * 4;
  float4 xv = *(const float4*)&xr[i0];
  float s = xv.x * xv.x + xv.y * xv.y + xv.z * xv.z + xv.w * xv.w;
#pragma unroll
  for (int off = 16; off; off >>= 1) s += __shfl_xor(s, off, 32);
  __shared__ float red[8];
  if (lane == 0) red[wid] = s;
  __syncthreads();
  if (tid == 0) {
    float tot = 0.f;
#pragma unroll
    for (int i = 0; i < 8; ++i) tot += red[i];
    red[0] = rsqrtf(tot / (float)Dm + 1e-5f);
  }
  __syncthreads();
  float sc = red[0];
  float4 wv = *(const float4*)&w[i0];
  unsigned p0 = (unsigned)f2bf(xv.x * sc * wv.x) | ((unsigned)f2bf(xv.y * sc * wv.y) << 16);
  unsigned p1 = (unsigned)f2bf(xv.z * sc * wv.z) | ((unsigned)f2bf(xv.w * sc * wv.w) << 16);
  uint2 pk; pk.x = p0; pk.y = p1;
  *(uint2*)&out[(size_t)t * Dm + i0] = pk;
}

// RoPE on (T, H, HD) f32 -> bf16
__global__ __launch_bounds__(256) void tb_rope_cast(const float* __restrict__ qin,
                                                    const float* __restrict__ cosf,
                                                    const float* __restrict__ sinf,
                                                    unsigned short* __restrict__ qout) {
  const int t = blockIdx.x;
  const int sidx = t & 1023;  // position within sequence (S = 1024)
#pragma unroll
  for (int pp = threadIdx.x; pp < 512; pp += 256) {
    int h = pp >> 5, j = pp & 31;
    size_t idx = (size_t)t * Dm + h * HDm + 2 * j;
    float xr = qin[idx], xi = qin[idx + 1];
    float c = cosf[sidx * 32 + j], sn = sinf[sidx * 32 + j];
    unsigned pk = (unsigned)f2bf(xr * c - xi * sn) |
                  ((unsigned)f2bf(xr * sn + xi * c) << 16);
    *(unsigned*)&qout[idx] = pk;
  }
}

// ---------- GEMM: C[M,N] = A[M,K](bf16) x Bt[K,N](bf16) (+resid) ----------
// block tile 64(M) x 256(N), wave tile 32x64, double-buffered async staging
template <bool HASRES>
__global__ __launch_bounds__(256) void tb_gemm_nn(const unsigned short* __restrict__ A,
                                                  const unsigned short* __restrict__ Bt,
                                                  float* __restrict__ C,
                                                  const float* __restrict__ resid,
                                                  int M, int N, int K) {
  const int tid = threadIdx.x, lane = tid & 31, wid = tid >> 5;
  const int wm = wid >> 2, wn = wid & 3;            // 2 x 4 waves
  const int m0 = blockIdx.y * 64, n0 = blockIdx.x * 256;
  __shared__ unsigned short As[2][64 * 40];
  __shared__ unsigned short Bs[2][32 * 264];
  v8f acc[2][4];
#pragma unroll
  for (int mi = 0; mi < 2; ++mi)
#pragma unroll
    for (int j = 0; j < 4; ++j) acc[mi][j] = zero8();

  const int ra = tid >> 2, ca = (tid & 3) * 8;      // A stage: 64x32
  const int rb = tid >> 5, cb = (tid & 31) * 8;     // B stage: 32x256, 8 rows apart
  auto issue = [&](int k, int buf) {
    const int k0 = k * 32;
    async_g2l_b128(&As[buf][ra * 40 + ca], &A[(size_t)(m0 + ra) * K + k0 + ca]);
#pragma unroll
    for (int q = 0; q < 4; ++q)
      async_g2l_b128(&Bs[buf][(rb + q * 8) * 264 + cb],
                     &Bt[(size_t)(k0 + rb + q * 8) * N + n0 + cb]);
  };

  const int nk = K >> 5;
  issue(0, 0);
  for (int k = 0; k < nk; ++k) {
    const int buf = k & 1;
    if (k + 1 < nk) { issue(k + 1, buf ^ 1); WAIT_ASYNC(5); }
    else            { WAIT_ASYNC(0); }
    __syncthreads();
    v16bf a0 = load_a_frag(&As[buf][(wm * 32) * 40], 40);
    v16bf a1 = load_a_frag(&As[buf][(wm * 32 + 16) * 40], 40);
#pragma unroll
    for (int j = 0; j < 4; ++j) {
      v16bf b = load_bN_frag(&Bs[buf][wn * 64 + j * 16], 264);
      acc[0][j] = wmma_bf16(a0, b, acc[0][j]);
      acc[1][j] = wmma_bf16(a1, b, acc[1][j]);
    }
    __syncthreads();
  }
  const int mr = ((lane >> 4) << 3), nc = lane & 15;
#pragma unroll
  for (int mi = 0; mi < 2; ++mi) {
#pragma unroll
    for (int r = 0; r < 8; ++r) {
      const size_t rowoff =
          (size_t)(m0 + wm * 32 + mi * 16 + mr + r) * N + n0 + wn * 64 + nc;
      float* Crow = C + rowoff;
#pragma unroll
      for (int j = 0; j < 4; ++j) {
        float v = acc[mi][j][r];
        if (HASRES) v += resid[rowoff + j * 16];
        Crow[j * 16] = v;
      }
    }
  }
}

// ---------- flash attention: per wave 16 q-rows, 32-key chunks, double-buffered ----------
__global__ __launch_bounds__(256) void tb_attention(const unsigned short* __restrict__ qb,
                                                    const unsigned short* __restrict__ kT,
                                                    const unsigned short* __restrict__ vb,
                                                    float* __restrict__ of) {
  const int tid = threadIdx.x, lane = tid & 31, wid = tid >> 5;
  const int y = blockIdx.x, h = blockIdx.y, b = blockIdx.z;
  const int bh = b * Hm + h;
  const int qtile = y * 128 + wid * 16;

  __shared__ unsigned short Kc[2][64 * 40];   // [hd][kv] (transposed K chunk)
  __shared__ unsigned short Vc[2][32 * 64];   // [kv][hd]
  __shared__ float Sc[8][512];                // per-wave 16x32 score scratch
  __shared__ float mArr[8][16], lArr[8][16], cArr[8][16];

  if (lane < 16) { mArr[wid][lane] = -1e30f; lArr[wid][lane] = 0.f; }

  const unsigned short* Qg = qb + (size_t)(b * 1024 + qtile) * Dm + h * HDm;
  v16bf qa0 = load_a_frag(Qg, Dm);
  v16bf qa1 = load_a_frag(Qg + 32, Dm);

  v8f oacc[4];
#pragma unroll
  for (int j = 0; j < 4; ++j) oacc[j] = zero8();

  const int rk = tid >> 2, ck = (tid & 3) * 8;   // Kc: hd row, kv cols
  const int rv = tid >> 3, cv = (tid & 7) * 8;   // Vc: kv row, hd cols
  auto issue = [&](int ch, int buf) {
    const int kv0 = ch * 32;
    async_g2l_b128(&Kc[buf][rk * 40 + ck],
                   &kT[(size_t)(bh * HDm + rk) * 1024 + kv0 + ck]);
    async_g2l_b128(&Vc[buf][rv * 64 + cv],
                   &vb[(size_t)(b * 1024 + kv0 + rv) * Dm + h * HDm + cv]);
  };

  const int nchunks = y * 4 + 4;
  issue(0, 0);
  for (int ch = 0; ch < nchunks; ++ch) {
    const int kv0 = ch * 32;
    const int buf = ch & 1;
    if (ch + 1 < nchunks) { issue(ch + 1, buf ^ 1); WAIT_ASYNC(2); }
    else                  { WAIT_ASYNC(0); }
    __syncthreads();
    const unsigned short* KcB = Kc[buf];
    const unsigned short* VcB = Vc[buf];
    const bool active = (kv0 <= qtile + 15);   // wave-uniform
    if (active) {
      v8f s0 = zero8(), s1 = zero8();
      s0 = wmma_bf16(qa0, load_bN_frag(&KcB[0 * 40 + 0], 40), s0);
      s0 = wmma_bf16(qa1, load_bN_frag(&KcB[32 * 40 + 0], 40), s0);
      s1 = wmma_bf16(qa0, load_bN_frag(&KcB[0 * 40 + 16], 40), s1);
      s1 = wmma_bf16(qa1, load_bN_frag(&KcB[32 * 40 + 16], 40), s1);
      const int mrB = ((lane >> 4) << 3), n = lane & 15;
#pragma unroll
      for (int r = 0; r < 8; ++r) {
        int m = mrB + r;
        int q = qtile + m;
        float v0 = s0[r] * 0.125f;
        float v1 = s1[r] * 0.125f;
        if (kv0 + n > q)       v0 = -1e30f;
        if (kv0 + 16 + n > q)  v1 = -1e30f;
        Sc[wid][m * 32 + n]      = v0;
        Sc[wid][m * 32 + 16 + n] = v1;
      }
    }
    wait_ds0();
    if (active && lane < 16) {
      const int m = lane;
      float mold = mArr[wid][m];
      float rmax = mold;
      float* srow = &Sc[wid][m * 32];
#pragma unroll
      for (int c2 = 0; c2 < 32; ++c2) rmax = fmaxf(rmax, srow[c2]);
      float corr = __expf(mold - rmax);
      float lsum = lArr[wid][m] * corr;
#pragma unroll
      for (int c2 = 0; c2 < 32; ++c2) {
        float p = __expf(srow[c2] - rmax);
        lsum += p;
        srow[c2] = p;
      }
      mArr[wid][m] = rmax; lArr[wid][m] = lsum; cArr[wid][m] = corr;
    }
    wait_ds0();
    if (active) {
      v16bf pa = load_a_frag_f32(&Sc[wid][0], 32);
      const int mrB = ((lane >> 4) << 3);
      float cr[8];
#pragma unroll
      for (int r = 0; r < 8; ++r) cr[r] = cArr[wid][mrB + r];
#pragma unroll
      for (int j = 0; j < 4; ++j) {
        v8f t = oacc[j];
#pragma unroll
        for (int r = 0; r < 8; ++r) t[r] *= cr[r];
        t = wmma_bf16(pa, load_bN_frag(&VcB[j * 16], 64), t);
        oacc[j] = t;
      }
    }
    __syncthreads();
  }
  wait_ds0();
  const int mrB = ((lane >> 4) << 3), nc = lane & 15;
  float inv[8];
#pragma unroll
  for (int r = 0; r < 8; ++r) inv[r] = 1.f / lArr[wid][mrB + r];
#pragma unroll
  for (int r = 0; r < 8; ++r) {
    float* orow = &of[(size_t)(b * 1024 + qtile + mrB + r) * Dm + h * HDm + nc];
#pragma unroll
    for (int j = 0; j < 4; ++j) orow[j * 16] = oacc[j][r] * inv[r];
  }
}

// ---------- MoE gate + top-2 ----------
__global__ __launch_bounds__(256) void tb_gate_topk(const unsigned short* __restrict__ tb,
                                                    const float* __restrict__ gw,
                                                    int* __restrict__ eidx,
                                                    float* __restrict__ ew,
                                                    int* __restrict__ counts) {
  const int lane = threadIdx.x & 31, wid = threadIdx.x >> 5;
  const int t = blockIdx.x * 8 + wid;
  float acc[Em];
#pragma unroll
  for (int e = 0; e < Em; ++e) acc[e] = 0.f;
  for (int d = lane; d < Dm; d += 32) {
    float xv = bf2f(tb[(size_t)t * Dm + d]);
#pragma unroll
    for (int e = 0; e < Em; ++e) acc[e] += xv * gw[e * Dm + d];
  }
#pragma unroll
  for (int e = 0; e < Em; ++e) {
#pragma unroll
    for (int off = 16; off; off >>= 1) acc[e] += __shfl_xor(acc[e], off, 32);
  }
  if (lane == 0) {
    int i0 = 0; float g0 = acc[0];
#pragma unroll
    for (int e = 1; e < Em; ++e) if (acc[e] > g0) { g0 = acc[e]; i0 = e; }
    int i1 = (i0 == 0) ? 1 : 0; float g1 = acc[i1];
#pragma unroll
    for (int e = 0; e < Em; ++e)
      if (e != i0 && acc[e] > g1) { g1 = acc[e]; i1 = e; }
    float e1 = __expf(g1 - g0);
    float w0 = 1.f / (1.f + e1);
    float w1 = e1 / (1.f + e1);
    eidx[2 * t] = i0; eidx[2 * t + 1] = i1;
    ew[2 * t] = w0;   ew[2 * t + 1] = w1;
    atomicAdd(&counts[i0], 1);
    atomicAdd(&counts[i1], 1);
  }
}

__global__ void tb_offsets(const int* __restrict__ counts, int* __restrict__ offs,
                           int* __restrict__ fill) {
  if (threadIdx.x == 0) {
    int o = 0;
    for (int e = 0; e < Em; ++e) {
      offs[e] = o;
      o += (counts[e] + 63) & ~63;
      fill[e] = 0;
    }
  }
}

__global__ __launch_bounds__(256) void tb_scatter(const int* __restrict__ eidx,
                                                  const float* __restrict__ ew,
                                                  const int* __restrict__ offs,
                                                  int* __restrict__ fill,
                                                  int* __restrict__ perm,
                                                  float* __restrict__ roww,
                                                  int* __restrict__ rowmap) {
  int t = blockIdx.x * 256 + threadIdx.x;
  if (t >= Tm) return;
#pragma unroll
  for (int k = 0; k < 2; ++k) {
    int e = eidx[2 * t + k];
    int slot = atomicAdd(&fill[e], 1);
    int grow = offs[e] + slot;
    perm[grow] = t;
    roww[grow] = ew[2 * t + k];
    rowmap[2 * t + k] = grow;
  }
}

// ---------- MoE: g = silu(t*W1^T) * (t*W3^T) * gate_weight -> bf16 ----------
// W1T/W3T pre-transposed per expert: (K=D, N=FF); double-buffered async B tiles
__global__ __launch_bounds__(256) void tb_moe_g(const unsigned short* __restrict__ tbuf,
                                                const unsigned short* __restrict__ w1T,
                                                const unsigned short* __restrict__ w3T,
                                                unsigned short* __restrict__ gbuf,
                                                const int* __restrict__ counts,
                                                const int* __restrict__ offs,
                                                const int* __restrict__ perm,
                                                const float* __restrict__ roww) {
  const int e = blockIdx.z;
  const int cnt = counts[e];
  const int mt = blockIdx.y;
  if (mt * 64 >= cnt) return;
  const int n0 = blockIdx.x * 128;
  const int base = offs[e];
  const int tid = threadIdx.x, lane = tid & 31, wid = tid >> 5;
  const int wm = wid >> 1, wn = wid & 1;

  __shared__ unsigned short As[2][64 * 40];
  __shared__ unsigned short B1s[2][32 * 136];
  __shared__ unsigned short B3s[2][32 * 136];
  __shared__ int tok[64];
  if (tid < 64) {
    int rl = mt * 64 + tid;
    tok[tid] = (rl < cnt) ? perm[base + rl] : -1;
  }
  __syncthreads();

  const unsigned short* W1 = w1T + (size_t)e * FFm * Dm;
  const unsigned short* W3 = w3T + (size_t)e * FFm * Dm;

  v8f a1[4], a3[4];
#pragma unroll
  for (int j = 0; j < 4; ++j) { a1[j] = zero8(); a3[j] = zero8(); }

  const int ra = tid >> 2, ca = (tid & 3) * 8;
  const int rb = tid >> 4, cb = (tid & 15) * 8;
  const int tka = tok[ra];
  auto issue = [&](int k, int buf) {
    const int k0 = k * 32;
    uint4 av; av.x = 0; av.y = 0; av.z = 0; av.w = 0;
    if (tka >= 0) av = *(const uint4*)&tbuf[(size_t)tka * Dm + k0 + ca];
    *(uint4*)&As[buf][ra * 40 + ca] = av;
    async_g2l_b128(&B1s[buf][rb * 136 + cb], &W1[(size_t)(k0 + rb) * FFm + n0 + cb]);
    async_g2l_b128(&B1s[buf][(rb + 16) * 136 + cb],
                   &W1[(size_t)(k0 + rb + 16) * FFm + n0 + cb]);
    async_g2l_b128(&B3s[buf][rb * 136 + cb], &W3[(size_t)(k0 + rb) * FFm + n0 + cb]);
    async_g2l_b128(&B3s[buf][(rb + 16) * 136 + cb],
                   &W3[(size_t)(k0 + rb + 16) * FFm + n0 + cb]);
  };

  const int nk = Dm >> 5;
  issue(0, 0);
  for (int k = 0; k < nk; ++k) {
    const int buf = k & 1;
    if (k + 1 < nk) { issue(k + 1, buf ^ 1); WAIT_ASYNC(4); }
    else            { WAIT_ASYNC(0); }
    __syncthreads();
    v16bf a = load_a_frag(&As[buf][(wm * 16) * 40], 40);
#pragma unroll
    for (int j = 0; j < 4; ++j) {
      v16bf b1 = load_bN_frag(&B1s[buf][wn * 64 + j * 16], 136);
      a1[j] = wmma_bf16(a, b1, a1[j]);
      v16bf b3 = load_bN_frag(&B3s[buf][wn * 64 + j * 16], 136);
      a3[j] = wmma_bf16(a, b3, a3[j]);
    }
    __syncthreads();
  }
  const int mr = ((lane >> 4) << 3), nc = lane & 15;
#pragma unroll
  for (int r = 0; r < 8; ++r) {
    const int rloc = mt * 64 + wm * 16 + mr + r;
    if (rloc < cnt) {
      const float wvr = roww[base + rloc];
      unsigned short* grow =
          &gbuf[(size_t)(base + rloc) * FFm + n0 + wn * 64 + nc];
#pragma unroll
      for (int j = 0; j < 4; ++j) {
        float x1 = a1[j][r], x3 = a3[j][r];
        float g = (x1 / (1.f + __expf(-x1))) * x3 * wvr;
        grow[j * 16] = f2bf(g);
      }
    }
  }
}

// ---------- MoE: mo[row, :] = g[row, :] @ W2[e]  (W2 already (K=FF, N=D)) ----------
// block tile 64 x 256, wave tile 32x64, double-buffered
__global__ __launch_bounds__(256) void tb_moe_out(const unsigned short* __restrict__ gbuf,
                                                  const unsigned short* __restrict__ w2b,
                                                  float* __restrict__ mo,
                                                  const int* __restrict__ counts,
                                                  const int* __restrict__ offs) {
  const int e = blockIdx.z;
  const int cnt = counts[e];
  const int mt = blockIdx.y;
  if (mt * 64 >= cnt) return;
  const int n0 = blockIdx.x * 256;
  const int base = offs[e];
  const int tid = threadIdx.x, lane = tid & 31, wid = tid >> 5;
  const int wm = wid >> 2, wn = wid & 3;

  __shared__ unsigned short As[2][64 * 40];
  __shared__ unsigned short Bs[2][32 * 264];
  const unsigned short* W2 = w2b + (size_t)e * FFm * Dm;

  v8f acc[2][4];
#pragma unroll
  for (int mi = 0; mi < 2; ++mi)
#pragma unroll
    for (int j = 0; j < 4; ++j) acc[mi][j] = zero8();

  const int ra = tid >> 2, ca = (tid & 3) * 8;
  const int rb = tid >> 5, cb = (tid & 31) * 8;
  const int rloca = mt * 64 + ra;
  auto issue = [&](int k, int buf) {
    const int k0 = k * 32;
    uint4 av; av.x = 0; av.y = 0; av.z = 0; av.w = 0;
    if (rloca < cnt) av = *(const uint4*)&gbuf[(size_t)(base + rloca) * FFm + k0 + ca];
    *(uint4*)&As[buf][ra * 40 + ca] = av;
#pragma unroll
    for (int q = 0; q < 4; ++q)
      async_g2l_b128(&Bs[buf][(rb + q * 8) * 264 + cb],
                     &W2[(size_t)(k0 + rb + q * 8) * Dm + n0 + cb]);
  };

  const int nk = FFm >> 5;
  issue(0, 0);
  for (int k = 0; k < nk; ++k) {
    const int buf = k & 1;
    if (k + 1 < nk) { issue(k + 1, buf ^ 1); WAIT_ASYNC(4); }
    else            { WAIT_ASYNC(0); }
    __syncthreads();
    v16bf a0 = load_a_frag(&As[buf][(wm * 32) * 40], 40);
    v16bf a1 = load_a_frag(&As[buf][(wm * 32 + 16) * 40], 40);
#pragma unroll
    for (int j = 0; j < 4; ++j) {
      v16bf b = load_bN_frag(&Bs[buf][wn * 64 + j * 16], 264);
      acc[0][j] = wmma_bf16(a0, b, acc[0][j]);
      acc[1][j] = wmma_bf16(a1, b, acc[1][j]);
    }
    __syncthreads();
  }
  const int mr = ((lane >> 4) << 3), nc = lane & 15;
#pragma unroll
  for (int mi = 0; mi < 2; ++mi) {
#pragma unroll
    for (int r = 0; r < 8; ++r) {
      const int rloc = mt * 64 + wm * 32 + mi * 16 + mr + r;
      if (rloc < cnt) {
        float* morow = &mo[(size_t)(base + rloc) * Dm + n0 + wn * 64 + nc];
#pragma unroll
        for (int j = 0; j < 4; ++j) morow[j * 16] = acc[mi][j][r];
      }
    }
  }
}

// ---------- out = h + moe(top2 rows)  (weights already folded into g) ----------
__global__ __launch_bounds__(256) void tb_final_add(const float* __restrict__ h,
                                                    const float* __restrict__ mo,
                                                    const int* __restrict__ rowmap,
                                                    float* __restrict__ out) {
  const int t = blockIdx.x;
  const int i0 = threadIdx.x * 4;
  const int r0 = rowmap[2 * t], r1 = rowmap[2 * t + 1];
  float4 hv = *(const float4*)&h[(size_t)t * Dm + i0];
  float4 m0 = *(const float4*)&mo[(size_t)r0 * Dm + i0];
  float4 m1 = *(const float4*)&mo[(size_t)r1 * Dm + i0];
  float4 ov;
  ov.x = hv.x + m0.x + m1.x;
  ov.y = hv.y + m0.y + m1.y;
  ov.z = hv.z + m0.z + m1.z;
  ov.w = hv.w + m0.w + m1.w;
  *(float4*)&out[(size_t)t * Dm + i0] = ov;
}

// ================= host launch =================
extern "C" void kernel_launch(void* const* d_in, const int* in_sizes, int n_in,
                              void* d_out, int out_size, void* d_ws, size_t ws_size,
                              hipStream_t stream) {
  (void)in_sizes; (void)n_in; (void)out_size; (void)ws_size;
  const float* x    = (const float*)d_in[0];
  const float* fcos = (const float*)d_in[1];
  const float* fsin = (const float*)d_in[2];
  const float* anw  = (const float*)d_in[3];
  const float* wq   = (const float*)d_in[4];
  const float* wk   = (const float*)d_in[5];
  const float* wv   = (const float*)d_in[6];
  const float* wo   = (const float*)d_in[7];
  const float* fnw  = (const float*)d_in[8];
  const float* gw   = (const float*)d_in[9];
  const float* w1   = (const float*)d_in[10];
  const float* w2   = (const float*)d_in[11];
  const float* w3   = (const float*)d_in[12];
  float* out = (float*)d_out;

  const size_t szDD  = (size_t)Dm * Dm;
  const size_t szEFD = (size_t)Em * FFm * Dm;
  const size_t szTD  = (size_t)Tm * Dm;

  char* wsb = (char*)d_ws;
  size_t off = 0;
  auto grab = [&](size_t bytes) -> char* {
    char* p = wsb + off;
    off = (off + bytes + 255) & ~(size_t)255;
    return p;
  };
  unsigned short* wqT = (unsigned short*)grab(szDD * 2);
  unsigned short* wkT = (unsigned short*)grab(szDD * 2);
  unsigned short* wvT = (unsigned short*)grab(szDD * 2);
  unsigned short* woT = (unsigned short*)grab(szDD * 2);
  unsigned short* w1T = (unsigned short*)grab(szEFD * 2);
  unsigned short* w2b = (unsigned short*)grab(szEFD * 2);
  unsigned short* w3T = (unsigned short*)grab(szEFD * 2);
  unsigned short* tbuf = (unsigned short*)grab(szTD * 2);
  float* qf = (float*)grab(szTD * 4);
  float* kf = (float*)grab(szTD * 4);
  float* vf = (float*)grab(szTD * 4);
  unsigned short* qbb = (unsigned short*)grab(szTD * 2);
  unsigned short* kbb = (unsigned short*)grab(szTD * 2);
  unsigned short* vbb = (unsigned short*)grab(szTD * 2);
  unsigned short* kTb = (unsigned short*)grab(szTD * 2);
  float* of = (float*)grab(szTD * 4);
  unsigned short* ob = (unsigned short*)grab(szTD * 2);
  float* hf = (float*)grab(szTD * 4);
  unsigned short* gbuf = (unsigned short*)grab((size_t)ROWSm * FFm * 2);
  float* mo = (float*)grab((size_t)ROWSm * Dm * 4);
  int* counts = (int*)grab(Em * 4);
  int* offs   = (int*)grab(Em * 4);
  int* fill   = (int*)grab(Em * 4);
  int* eidx   = (int*)grab((size_t)Tm * 2 * 4);
  float* ewt  = (float*)grab((size_t)Tm * 2 * 4);
  int* perm   = (int*)grab((size_t)ROWSm * 4);
  float* roww = (float*)grab((size_t)ROWSm * 4);
  int* rowmap = (int*)grab((size_t)Tm * 2 * 4);

  // 0) zero the expert counters (ws is poisoned, never re-poisoned)
  tb_zero_ints<<<1, 32, 0, stream>>>(counts, Em);

  // 1) one-time bf16 weight casts; projection weights and w1/w3 transposed to (K,N)
  tb_castT<<<dim3(32, 32, 1), 256, 0, stream>>>(wq, wqT, Dm, Dm);
  tb_castT<<<dim3(32, 32, 1), 256, 0, stream>>>(wk, wkT, Dm, Dm);
  tb_castT<<<dim3(32, 32, 1), 256, 0, stream>>>(wv, wvT, Dm, Dm);
  tb_castT<<<dim3(32, 32, 1), 256, 0, stream>>>(wo, woT, Dm, Dm);
  tb_castT<<<dim3(32, FFm / 32, Em), 256, 0, stream>>>(w1, w1T, FFm, Dm);
  tb_castT<<<dim3(32, FFm / 32, Em), 256, 0, stream>>>(w3, w3T, FFm, Dm);
  tb_cast8<<<(unsigned)(szEFD / 2048), 256, 0, stream>>>(w2, w2b, szEFD);

  // 2) attention rmsnorm -> bf16 activations
  tb_rmsnorm_cast<<<Tm, 256, 0, stream>>>(x, anw, tbuf);

  // 3) Q,K,V projections (WMMA NN gemm with pre-transposed weights)
  dim3 gproj(Dm / 256, Tm / 64);
  tb_gemm_nn<false><<<gproj, 256, 0, stream>>>(tbuf, wqT, qf, nullptr, Tm, Dm, Dm);
  tb_gemm_nn<false><<<gproj, 256, 0, stream>>>(tbuf, wkT, kf, nullptr, Tm, Dm, Dm);
  tb_gemm_nn<false><<<gproj, 256, 0, stream>>>(tbuf, wvT, vf, nullptr, Tm, Dm, Dm);

  // 4) RoPE + bf16 casts; K additionally transposed to (b,h,hd,s) for attention
  tb_rope_cast<<<Tm, 256, 0, stream>>>(qf, fcos, fsin, qbb);
  tb_rope_cast<<<Tm, 256, 0, stream>>>(kf, fcos, fsin, kbb);
  tb_cast8<<<(unsigned)(szTD / 2048), 256, 0, stream>>>(vf, vbb, szTD);
  tb_ktrans<<<dim3(32, 2, 32), 256, 0, stream>>>(kbb, kTb);

  // 5) flash attention
  tb_attention<<<dim3(8, Hm, 2), 256, 0, stream>>>(qbb, kTb, vbb, of);

  // 6) output projection + residual: h = x + attn @ wo^T
  tb_cast8<<<(unsigned)(szTD / 2048), 256, 0, stream>>>(of, ob, szTD);
  tb_gemm_nn<true><<<gproj, 256, 0, stream>>>(ob, woT, hf, x, Tm, Dm, Dm);

  // 7) ffn rmsnorm -> bf16 (reuse tbuf)
  tb_rmsnorm_cast<<<Tm, 256, 0, stream>>>(hf, fnw, tbuf);

  // 8) gate + top-2 + expert compaction
  tb_gate_topk<<<Tm / 8, 256, 0, stream>>>(tbuf, gw, eidx, ewt, counts);
  tb_offsets<<<1, 32, 0, stream>>>(counts, offs, fill);
  tb_scatter<<<Tm / 256, 256, 0, stream>>>(eidx, ewt, offs, fill, perm, roww, rowmap);

  // 9) expert MLPs on compacted rows only (top-2 per token)
  tb_moe_g<<<dim3(FFm / 128, 64, Em), 256, 0, stream>>>(tbuf, w1T, w3T, gbuf,
                                                        counts, offs, perm, roww);
  tb_moe_out<<<dim3(Dm / 256, 64, Em), 256, 0, stream>>>(gbuf, w2b, mo, counts, offs);

  // 10) out = h + moe
  tb_final_add<<<Tm, 256, 0, stream>>>(hf, mo, rowmap, out);
}
